// SelfMultiheadAttention_49435073577655
// MI455X (gfx1250) — matrix operand order repaired and checked by reading, not compiled, not run on hardware
//
#include <hip/hip_runtime.h>
#include <hip/hip_bf16.h>
#include <math.h>

// ---------------------------------------------------------------------------
// Self multi-head attention for MI455X (gfx1250, wave32, WMMA).
// f16 WMMA (16x16x32) with f32 accumulation; flash-style online softmax so the
// 48x1024x1024 score matrix never hits HBM; attn_bias (192MB) is the bandwidth
// floor and is streamed coalesced + prefetched; K-tile staging uses the CDNA5
// async global->LDS path (ASYNCcnt); GEMMs are 2x2 register-blocked per wave
// to double the FLOP/byte drawn from L2.
// ---------------------------------------------------------------------------

typedef __attribute__((ext_vector_type(16))) _Float16 v16h;
typedef __attribute__((ext_vector_type(8)))  _Float16 v8h;
typedef __attribute__((ext_vector_type(8)))  float    v8f;

#define N_NODE   1024
#define NGRAPH   4
#define EDIM     768
#define NHEADS   12
#define BHDIM    48        // NGRAPH*NHEADS
#define HD       64
#define M_ROWS   4096      // N_NODE*NGRAPH  (row m = n*NGRAPH + b)
#define QKV_COLS 2304      // 3*EDIM

// --- WMMA fragment gathers (contiguous k per half-wave, A and B alike) ---------------
// A 16x32 f16: lane l -> row (l&15); halves 0..15 = K off..off+15,
//              off = (l>>4)*16.
__device__ __forceinline__ v16h load_a_g(const _Float16* base, int ld,
                                         int m0, int k0, int lane) {
  int row = m0 + (lane & 15);
  int off = ((lane >> 4) & 1) * 16;
  const _Float16* p = base + (size_t)row * ld + k0 + off;
  return *(const v16h*)(p);
}

// B 32x16 f16 with B[k][n] = W[n][k] (W row-major, rows contiguous over k):
// lane l -> column (l&15) = row of W; 16 contiguous K at k0 + (l>>4)*16.
__device__ __forceinline__ v16h load_b_g(const _Float16* base, int ld,
                                         int n0, int k0, int lane) {
  int row  = n0 + (lane & 15);
  int koff = k0 + ((lane >> 4) & 1) * 16;
  return *(const v16h*)(base + (size_t)row * ld + koff);
}

// --- f32 -> f16 conversion of query / W_in / W_out --------------------------
__global__ __launch_bounds__(256) void convert_f16_kernel(
    const float* __restrict__ q, const float* __restrict__ win,
    const float* __restrict__ wout, _Float16* __restrict__ Xh,
    _Float16* __restrict__ Winh, _Float16* __restrict__ Wouth) {
  const int NA = M_ROWS * EDIM;
  const int NB = QKV_COLS * EDIM;
  const int NC = EDIM * EDIM;
  int total = NA + NB + NC;
  for (int i = blockIdx.x * blockDim.x + threadIdx.x; i < total;
       i += gridDim.x * blockDim.x) {
    if (i < NA)             Xh[i] = (_Float16)q[i];
    else if (i < NA + NB)   { int j = i - NA;       Winh[j]  = (_Float16)win[j];  }
    else                    { int j = i - NA - NB;  Wouth[j] = (_Float16)wout[j]; }
  }
}

// --- QKV projection: [4096 x 2304] = Xh @ W_in^T + b_in ---------------------
// One wave per 32x32 output tile (2x2 WMMA fragments -> 16 FLOP/B from L2).
// Writes f16 Q (pre-scaled), K, V in [BH, N, HD] layout.
__device__ __forceinline__ void qkv_store_tile(
    const v8f& acc, int m0, int j0, int lane, const float* __restrict__ b_in,
    _Float16* __restrict__ Qh, _Float16* __restrict__ Kh,
    _Float16* __restrict__ Vh) {
  int col   = j0 + (lane & 15);
  int which = col / EDIM;                       // 0=q 1=k 2=v
  int e = col % EDIM;
  int h = e / HD, d = e % HD;
  float bias = b_in[col];
  float sc   = (which == 0) ? 0.125f : 1.0f;    // hd^-0.5, applied after bias
  _Float16* dst = (which == 0) ? Qh : (which == 1 ? Kh : Vh);
  int rbase = m0 + ((lane >> 4) & 1) * 8;
#pragma unroll
  for (int r = 0; r < 8; ++r) {
    int m = rbase + r;
    int n = m >> 2, bb = m & 3;                 // m = n*NGRAPH + b
    float v = (acc[r] + bias) * sc;
    dst[(((size_t)(bb * NHEADS + h)) * N_NODE + n) * HD + d] = (_Float16)v;
  }
}

__global__ __launch_bounds__(256) void qkv_gemm_kernel(
    const _Float16* __restrict__ Xh, const _Float16* __restrict__ Wh,
    const float* __restrict__ b_in, _Float16* __restrict__ Qh,
    _Float16* __restrict__ Kh, _Float16* __restrict__ Vh) {
  int wave = blockIdx.x * 8 + (threadIdx.x >> 5);
  int lane = threadIdx.x & 31;
  const int NT = QKV_COLS / 32;                 // 72 column tiles
  int mt = wave / NT, nt = wave % NT;
  int m0 = mt * 32, j0 = nt * 32;

  v8f a00 = {}, a01 = {}, a10 = {}, a11 = {};
#pragma unroll 2
  for (int k0 = 0; k0 < EDIM; k0 += 32) {
    v16h fa0 = load_a_g(Xh, EDIM, m0,      k0, lane);
    v16h fa1 = load_a_g(Xh, EDIM, m0 + 16, k0, lane);
    v16h fb0 = load_b_g(Wh, EDIM, j0,      k0, lane);
    v16h fb1 = load_b_g(Wh, EDIM, j0 + 16, k0, lane);
    a00 = __builtin_amdgcn_wmma_f32_16x16x32_f16(false, fa0, false, fb0,
                                                 (short)0, a00, false, false);
    a01 = __builtin_amdgcn_wmma_f32_16x16x32_f16(false, fa0, false, fb1,
                                                 (short)0, a01, false, false);
    a10 = __builtin_amdgcn_wmma_f32_16x16x32_f16(false, fa1, false, fb0,
                                                 (short)0, a10, false, false);
    a11 = __builtin_amdgcn_wmma_f32_16x16x32_f16(false, fa1, false, fb1,
                                                 (short)0, a11, false, false);
  }
  qkv_store_tile(a00, m0,      j0,      lane, b_in, Qh, Kh, Vh);
  qkv_store_tile(a01, m0,      j0 + 16, lane, b_in, Qh, Kh, Vh);
  qkv_store_tile(a10, m0 + 16, j0,      lane, b_in, Qh, Kh, Vh);
  qkv_store_tile(a11, m0 + 16, j0 + 16, lane, b_in, Qh, Kh, Vh);
}

// --- Flash attention: block = one (b,h) x 128 query rows, 8 waves x 16 rows -
__global__ __launch_bounds__(256) void attn_kernel(
    const _Float16* __restrict__ Qh, const _Float16* __restrict__ Kh,
    const _Float16* __restrict__ Vh, const float* __restrict__ bias,
    _Float16* __restrict__ Ctx) {
  __shared__ __align__(32) _Float16 sK[32][64];      // 32 keys x 64 hd
  __shared__ __align__(32) _Float16 sVt[64][32];     // transposed V tile
  __shared__ __align__(32) _Float16 sP[8][16][32];   // per-wave P scratch

  int bh   = blockIdx.x >> 3;                   // 48 head-batches
  int qblk = blockIdx.x & 7;                    // 8 blocks of 128 rows
  int w    = threadIdx.x >> 5;
  int lane = threadIdx.x & 31;
  int q0 = qblk * 128 + w * 16;
  int b = bh / NHEADS, h = bh % NHEADS;

  const _Float16* Qb = Qh + (size_t)bh * N_NODE * HD;
  const _Float16* Kb = Kh + (size_t)bh * N_NODE * HD;
  const _Float16* Vb = Vh + (size_t)bh * N_NODE * HD;
  const float*    Bb = bias + (size_t)bh * N_NODE * N_NODE;

  // Q fragments (reduction dim hd=64 -> two K=32 chunks), reused all tiles.
  v16h aq0 = load_a_g(Qb, HD, q0, 0, lane);
  v16h aq1 = load_a_g(Qb, HD, q0, 32, lane);

  int halfsel = (lane >> 4) & 1;
  int rbase   = halfsel * 8;                    // rows this half-wave owns
  int cbase   = lane & 15;                      // column within tile

  // Per-thread staging coordinates (32 keys x 64 hd tile, 8 halves each).
  int sflat = threadIdx.x * 8;
  int skey = sflat >> 6, shd = sflat & 63;
  unsigned int sK_dst = (unsigned int)(uintptr_t)&sK[skey][shd];

  float mrow[8], lrow[8];
  v8f o0 = {}, o1 = {}, o2 = {}, o3 = {};       // 16 x 64 f32 accumulator
#pragma unroll
  for (int r = 0; r < 8; ++r) { mrow[r] = -1e30f; lrow[r] = 0.0f; }

  for (int kt = 0; kt < N_NODE; kt += 32) {
    __syncthreads();                            // prior readers of sK/sVt done
    { // K tile: CDNA5 async global->LDS copy (no VGPR round trip, ASYNCcnt)
      const _Float16* gk = Kb + (size_t)(kt + skey) * HD + shd;
      asm volatile("global_load_async_to_lds_b128 %0, %1, off"
                   :: "v"(sK_dst), "v"(gk) : "memory");
      // V tile: transpose needs half-granularity scatter -> via VGPRs
      v8h vv = *(const v8h*)(Vb + (size_t)(kt + skey) * HD + shd);
#pragma unroll
      for (int i = 0; i < 8; ++i) sVt[shd + i][skey] = vv[i];
    }
    asm volatile("s_wait_asynccnt 0" ::: "memory");
    __syncthreads();

    // Prefetch next attn_bias tile (one aligned 128B line per row).
    if (kt + 32 < N_NODE) {
#pragma unroll
      for (int r = 0; r < 8; ++r)
        __builtin_prefetch(Bb + (size_t)(q0 + rbase + r) * N_NODE + kt + 32,
                           0, 0);
    }

    // S = Q K^T : B[k=hd][n=key] = sK[key][hd] (contiguous over hd)
    v8f s0 = {}, s1 = {};
    {
      v16h bk00 = *(const v16h*)(&sK[cbase][halfsel * 16]);
      v16h bk01 = *(const v16h*)(&sK[cbase][32 + halfsel * 16]);
      v16h bk10 = *(const v16h*)(&sK[16 + cbase][halfsel * 16]);
      v16h bk11 = *(const v16h*)(&sK[16 + cbase][32 + halfsel * 16]);
      s0 = __builtin_amdgcn_wmma_f32_16x16x32_f16(false, aq0, false, bk00,
                                                  (short)0, s0, false, false);
      s0 = __builtin_amdgcn_wmma_f32_16x16x32_f16(false, aq1, false, bk01,
                                                  (short)0, s0, false, false);
      s1 = __builtin_amdgcn_wmma_f32_16x16x32_f16(false, aq0, false, bk10,
                                                  (short)0, s1, false, false);
      s1 = __builtin_amdgcn_wmma_f32_16x16x32_f16(false, aq1, false, bk11,
                                                  (short)0, s1, false, false);
    }

    // Stream attn_bias (coalesced 64B per half-wave row) and online softmax.
#pragma unroll
    for (int r = 0; r < 8; ++r) {
      const float* bp = Bb + (size_t)(q0 + rbase + r) * N_NODE + kt;
      s0[r] += bp[cbase];
      s1[r] += bp[16 + cbase];
    }
#pragma unroll
    for (int r = 0; r < 8; ++r) {
      float t = fmaxf(s0[r], s1[r]);
#pragma unroll
      for (int xm = 8; xm >= 1; xm >>= 1) t = fmaxf(t, __shfl_xor(t, xm, 32));
      float mn    = fmaxf(mrow[r], t);
      float alpha = __expf(mrow[r] - mn);
      float p0 = __expf(s0[r] - mn);
      float p1 = __expf(s1[r] - mn);
      float ls = p0 + p1;
#pragma unroll
      for (int xm = 8; xm >= 1; xm >>= 1) ls += __shfl_xor(ls, xm, 32);
      lrow[r] = lrow[r] * alpha + ls;
      mrow[r] = mn;
      o0[r] *= alpha; o1[r] *= alpha; o2[r] *= alpha; o3[r] *= alpha;
      sP[w][rbase + r][cbase]      = (_Float16)p0;   // D-layout -> LDS
      sP[w][rbase + r][16 + cbase] = (_Float16)p1;
    }
    // same-wave LDS RAW across lanes: drain DS before A-layout re-gather
    asm volatile("s_wait_dscnt 0" ::: "memory");

    // P@V : A = P (16x32 keys), B[k=key][n=hd] = sVt[hd][key]
    v16h ap;
    {
      int row = lane & 15, off = halfsel * 16;
      ap = *(const v16h*)(&sP[w][row][off]);
    }
    v16h bv0 = *(const v16h*)(&sVt[ 0 + cbase][halfsel * 16]);
    v16h bv1 = *(const v16h*)(&sVt[16 + cbase][halfsel * 16]);
    v16h bv2 = *(const v16h*)(&sVt[32 + cbase][halfsel * 16]);
    v16h bv3 = *(const v16h*)(&sVt[48 + cbase][halfsel * 16]);
    o0 = __builtin_amdgcn_wmma_f32_16x16x32_f16(false, ap, false, bv0,
                                                (short)0, o0, false, false);
    o1 = __builtin_amdgcn_wmma_f32_16x16x32_f16(false, ap, false, bv1,
                                                (short)0, o1, false, false);
    o2 = __builtin_amdgcn_wmma_f32_16x16x32_f16(false, ap, false, bv2,
                                                (short)0, o2, false, false);
    o3 = __builtin_amdgcn_wmma_f32_16x16x32_f16(false, ap, false, bv3,
                                                (short)0, o3, false, false);
  }

  // Normalize and write context as f16 [N*B, E] for the output GEMM.
#pragma unroll
  for (int r = 0; r < 8; ++r) {
    float inv = 1.0f / lrow[r];
    int n = q0 + rbase + r;
    size_t base = ((size_t)n * NGRAPH + b) * EDIM + (size_t)h * HD;
    Ctx[base +  0 + cbase] = (_Float16)(o0[r] * inv);
    Ctx[base + 16 + cbase] = (_Float16)(o1[r] * inv);
    Ctx[base + 32 + cbase] = (_Float16)(o2[r] * inv);
    Ctx[base + 48 + cbase] = (_Float16)(o3[r] * inv);
  }
}

// --- Output projection: out = Ctx @ W_out^T + b_out, masked -----------------
__device__ __forceinline__ void out_store_tile(
    const v8f& acc, int m0, int j0, int lane, const float* __restrict__ b_out,
    const unsigned char* __restrict__ mask, float* __restrict__ out) {
  int col = j0 + (lane & 15);
  float bias = b_out[col];
  int rbase = m0 + ((lane >> 4) & 1) * 8;
#pragma unroll
  for (int r = 0; r < 8; ++r) {
    int m = rbase + r;
    int n = m >> 2, bb = m & 3;                 // mask[b][n]
    float v = acc[r] + bias;
    bool keep = mask[bb * N_NODE + n] != 0;
    out[(size_t)m * EDIM + col] = keep ? v : 0.0f;
  }
}

__global__ __launch_bounds__(256) void out_gemm_kernel(
    const _Float16* __restrict__ Ctx, const _Float16* __restrict__ Wh,
    const float* __restrict__ b_out, const unsigned char* __restrict__ mask,
    float* __restrict__ out) {
  int wave = blockIdx.x * 8 + (threadIdx.x >> 5);
  int lane = threadIdx.x & 31;
  const int NT = EDIM / 32;                     // 24 column tiles
  int mt = wave / NT, nt = wave % NT;
  int m0 = mt * 32, j0 = nt * 32;

  v8f a00 = {}, a01 = {}, a10 = {}, a11 = {};
#pragma unroll 2
  for (int k0 = 0; k0 < EDIM; k0 += 32) {
    v16h fa0 = load_a_g(Ctx, EDIM, m0,      k0, lane);
    v16h fa1 = load_a_g(Ctx, EDIM, m0 + 16, k0, lane);
    v16h fb0 = load_b_g(Wh, EDIM, j0,      k0, lane);
    v16h fb1 = load_b_g(Wh, EDIM, j0 + 16, k0, lane);
    a00 = __builtin_amdgcn_wmma_f32_16x16x32_f16(false, fa0, false, fb0,
                                                 (short)0, a00, false, false);
    a01 = __builtin_amdgcn_wmma_f32_16x16x32_f16(false, fa0, false, fb1,
                                                 (short)0, a01, false, false);
    a10 = __builtin_amdgcn_wmma_f32_16x16x32_f16(false, fa1, false, fb0,
                                                 (short)0, a10, false, false);
    a11 = __builtin_amdgcn_wmma_f32_16x16x32_f16(false, fa1, false, fb1,
                                                 (short)0, a11, false, false);
  }
  out_store_tile(a00, m0,      j0,      lane, b_out, mask, out);
  out_store_tile(a01, m0,      j0 + 16, lane, b_out, mask, out);
  out_store_tile(a10, m0 + 16, j0,      lane, b_out, mask, out);
  out_store_tile(a11, m0 + 16, j0 + 16, lane, b_out, mask, out);
}

// ---------------------------------------------------------------------------
extern "C" void kernel_launch(void* const* d_in, const int* in_sizes, int n_in,
                              void* d_out, int out_size, void* d_ws,
                              size_t ws_size, hipStream_t stream) {
  (void)in_sizes; (void)n_in; (void)out_size; (void)ws_size;
  const float*         query = (const float*)d_in[0];
  const float*         abias = (const float*)d_in[1];
  const unsigned char* nmask = (const unsigned char*)d_in[2]; // jax bool = 1B
  const float*         W_in  = (const float*)d_in[3];
  const float*         b_in  = (const float*)d_in[4];
  const float*         W_out = (const float*)d_in[5];
  const float*         b_out = (const float*)d_in[6];
  float*               out   = (float*)d_out;

  // Carve f16 scratch from d_ws (256B-aligned sub-allocations, ~36MB total).
  _Float16* p = (_Float16*)d_ws;
  auto carve = [&](size_t n) {
    _Float16* r = p;
    p += ((n + (size_t)127) & ~(size_t)127);
    return r;
  };
  _Float16* Xh    = carve((size_t)M_ROWS * EDIM);
  _Float16* Winh  = carve((size_t)QKV_COLS * EDIM);
  _Float16* Wouth = carve((size_t)EDIM * EDIM);
  _Float16* Qh    = carve((size_t)BHDIM * N_NODE * HD);
  _Float16* Kh    = carve((size_t)BHDIM * N_NODE * HD);
  _Float16* Vh    = carve((size_t)BHDIM * N_NODE * HD);
  _Float16* Ctx   = carve((size_t)M_ROWS * EDIM);

  const int cvt_total = M_ROWS * EDIM + QKV_COLS * EDIM + EDIM * EDIM;
  convert_f16_kernel<<<(cvt_total + 255) / 256, 256, 0, stream>>>(
      query, W_in, W_out, Xh, Winh, Wouth);

  // 128 row-tiles x 72 col-tiles (32x32 per wave), 8 waves per block.
  qkv_gemm_kernel<<<(128 * 72) / 8, 256, 0, stream>>>(
      Xh, Winh, b_in, Qh, Kh, Vh);

  // 48 head-batches x 8 query blocks of 128 rows.
  attn_kernel<<<BHDIM * 8, 256, 0, stream>>>(Qh, Kh, Vh, abias, Ctx);

  // 128 row-tiles x 24 col-tiles (32x32 per wave).
  out_gemm_kernel<<<(128 * 24) / 8, 256, 0, stream>>>(
      Ctx, Wouth, b_out, nmask, out);
}
